// CrossAttentionAggregator_57423712748008
// MI455X (gfx1250) — matrix operand order repaired and checked by reading, not compiled, not run on hardware
//
#include <hip/hip_runtime.h>
#include <hip/hip_bf16.h>
#include <math.h>

#define B_  32
#define C_  64
#define L_  4096
#define ED  384
#define DV  128
#define NH  4
#define HD  32
#define LN_EPS 1e-5f

typedef __attribute__((ext_vector_type(16))) _Float16 v16h;
typedef __attribute__((ext_vector_type(8)))  _Float16 v8h;
typedef __attribute__((ext_vector_type(8)))  float    v8f;

__device__ __forceinline__ void cvt4(v16h& v, int base, float4 f) {
  v[base + 0] = (_Float16)f.x; v[base + 1] = (_Float16)f.y;
  v[base + 2] = (_Float16)f.z; v[base + 3] = (_Float16)f.w;
}

// ---------------------------------------------------------------------------
// Kernel 1: per-(b,c) softmax row stats.
//   rowmax[(b*C+c)*NH+h] = max_{l valid} s[l]/tau_h
//   rowsum[(b*C+c)*NH+h] = sum_{l valid} exp(s[l]/tau_h - rowmax)
// ---------------------------------------------------------------------------
__global__ __launch_bounds__(256)
void stats_kernel(const float* __restrict__ scores,
                  const unsigned char* __restrict__ mask,
                  const float* __restrict__ log_tau,
                  float* __restrict__ rowmax,
                  float* __restrict__ rowsum)
{
  const int row  = blockIdx.x;          // b*C + c
  const int b    = row / C_;
  const int t    = threadIdx.x;
  const int wave = t >> 5;
  const int lane = t & 31;

  __shared__ float redmax[8];
  __shared__ float redsum[NH][8];

  const float* srow = scores + (size_t)row * L_;
  const unsigned char* mrow = mask + (size_t)b * L_;

  float s[16];
  bool  pad[16];
  float mx = -3.402823466e38f;
  #pragma unroll
  for (int i = 0; i < 16; ++i) {
    const int l = t + 256 * i;
    s[i]   = srow[l];
    pad[i] = (mrow[l] != 0);
    if (!pad[i]) mx = fmaxf(mx, s[i]);
  }
  #pragma unroll
  for (int off = 16; off; off >>= 1)
    mx = fmaxf(mx, __shfl_xor(mx, off, 32));
  if (lane == 0) redmax[wave] = mx;
  __syncthreads();
  float gmax = redmax[0];
  #pragma unroll
  for (int w = 1; w < 8; ++w) gmax = fmaxf(gmax, redmax[w]);

  #pragma unroll
  for (int h = 0; h < NH; ++h) {
    const float itau = __expf(-log_tau[h]);   // 1 / exp(log_tau)
    const float mh   = gmax * itau;
    float acc = 0.f;
    #pragma unroll
    for (int i = 0; i < 16; ++i)
      if (!pad[i]) acc += __expf(s[i] * itau - mh);
    #pragma unroll
    for (int off = 16; off; off >>= 1)
      acc += __shfl_xor(acc, off, 32);
    if (lane == 0) redsum[h][wave] = acc;
  }
  __syncthreads();
  if (t < NH) {
    float tot = 0.f;
    #pragma unroll
    for (int w = 0; w < 8; ++w) tot += redsum[t][w];
    const float itau = __expf(-log_tau[t]);
    rowmax[(size_t)row * NH + t] = gmax * itau;
    rowsum[(size_t)row * NH + t] = tot;
  }
}

// ---------------------------------------------------------------------------
// Kernel 2: V projection, V = emb @ w_v^T, stored transposed in f16:
//   VT[b][d][l],  d in [0,128), l in [0,4096)
// One block = one (b, l-tile of 16); 8 waves = 8 d-tiles of 16.
// ---------------------------------------------------------------------------
__global__ __launch_bounds__(256)
void vproj_kernel(const float* __restrict__ emb,   // (B, L, ED)
                  const float* __restrict__ w_v,   // (DV, ED)
                  _Float16* __restrict__ VT)       // (B, DV, L)
{
  const int ltile = blockIdx.x % (L_ / 16);
  const int b     = blockIdx.x / (L_ / 16);
  const int wave  = threadIdx.x >> 5;
  const int lane  = threadIdx.x & 31;
  const int half  = lane >> 4;
  const int lan   = lane & 15;

  const int l0 = ltile * 16;
  const int n0 = wave * 16;

  const int akb = half * 8;      // A-operand K base within 32-chunk
  const int bkb = half * 8;      // B-operand K base within 32-chunk

  const float* Abase = emb + ((size_t)b * L_ + (l0 + lan)) * ED;
  const float* Bbase = w_v + (size_t)(n0 + lan) * ED;

  v8f acc = {};
  #pragma unroll
  for (int k0 = 0; k0 < ED; k0 += 32) {
    v16h a, bm;
    const float4* ap = (const float4*)(Abase + k0 + akb);
    cvt4(a, 0, ap[0]); cvt4(a, 4, ap[1]);              // K local 0..7
    const float4* ap2 = (const float4*)(Abase + k0 + akb + 16);
    cvt4(a, 8, ap2[0]); cvt4(a, 12, ap2[1]);           // K local 16..23

    const float4* bp = (const float4*)(Bbase + k0 + bkb);
    cvt4(bm, 0, bp[0]); cvt4(bm, 4, bp[1]);
    cvt4(bm, 8, bp[4]); cvt4(bm, 12, bp[5]);

    acc = __builtin_amdgcn_wmma_f32_16x16x32_f16(
        false, a, false, bm, (short)0, acc, false, false);
  }

  // D tile: lane -> column d = n0+lan, VGPR j -> row l = l0 + half*8 + j
  _Float16* outp = VT + ((size_t)b * DV + (n0 + lan)) * L_ + l0 + half * 8;
  v8h o;
  #pragma unroll
  for (int j = 0; j < 8; ++j) o[j] = (_Float16)acc[j];
  *(v8h*)outp = o;
}

// ---------------------------------------------------------------------------
// Kernel 3: flash-style attention GEMM with async-to-LDS B-operand staging.
// Block = (b, h, d-tile of 16); 4 waves = 4 c-tiles of 16. K loop over L.
// All 4 waves share the same B tile VT[b][d0..d0+15][l0..l0+31] (1 KB), so it
// is DMA'd into LDS once per K-step (double buffered, ASYNCcnt-tracked) and
// consumed via ds_load, overlapping with the v_exp_f32 weight generation.
// ---------------------------------------------------------------------------
__global__ __launch_bounds__(128)
void attn_kernel(const float* __restrict__ scores,
                 const unsigned char* __restrict__ mask,
                 const float* __restrict__ log_tau,
                 const _Float16* __restrict__ VT,
                 const float* __restrict__ rowmax,
                 const float* __restrict__ rowsum,
                 _Float16* __restrict__ attn_out)
{
  const int blk  = blockIdx.x;          // b*NH*2 + h*2 + dt
  const int dt   = blk & 1;
  const int h    = (blk >> 1) & (NH - 1);
  const int b    = blk >> 3;
  const int tid  = threadIdx.x;
  const int wave = tid >> 5;            // c-tile
  const int lane = tid & 31;
  const int half = lane >> 4;
  const int lan  = lane & 15;

  const int c0 = wave * 16;
  const int d0 = h * HD + dt * 16;
  const int akb = half * 16;
  const int bkb = half * 16;

  // 2 x (16 rows x 32 halves) double buffer, 1 KB each
  __shared__ _Float16 sbuf[2][16 * 32];

  const float itau = __expf(-log_tau[h]);
  const int   arow = c0 + lan;
  const float mrow = rowmax[(size_t)(b * C_ + arow) * NH + h];
  const float* srow = scores + ((size_t)b * C_ + arow) * L_;
  const unsigned char* mbase = mask + (size_t)b * L_;

  // async staging: thread -> (row = tid>>3, 8-byte segment = tid&7)
  const int srow16 = tid >> 3;
  const int sseg   = tid & 7;
  const _Float16* vsrc =
      VT + ((size_t)b * DV + (d0 + srow16)) * L_ + sseg * 4;
  const unsigned ldsoff =
      (unsigned)(uintptr_t)(&sbuf[0][0]) + srow16 * 64 + sseg * 8;

  float rs[8];
  #pragma unroll
  for (int j = 0; j < 8; ++j)
    rs[j] = 1.0f / rowsum[(size_t)(b * C_ + c0 + half * 8 + j) * NH + h];

  // prologue: fill buffer 0 with the l0 = 0 tile
  {
    unsigned dst = ldsoff;
    unsigned long long src = (unsigned long long)(uintptr_t)vsrc;
    asm volatile("global_load_async_to_lds_b64 %0, %1, off"
                 :: "v"(dst), "v"(src) : "memory");
  }

  v8f acc = {};
  int cur = 0;
  for (int l0 = 0; l0 < L_; l0 += 32) {
    if (l0 + 256 < L_)
      __builtin_prefetch(srow + l0 + 256, 0, 1);

    // ---- A operand: unnormalized softmax weights, computed on the fly ----
    v16h a;
    {
      const float4* sp = (const float4*)(srow + l0 + akb);
      float4 f0 = sp[0], f1 = sp[1];
      unsigned long long mb =
          *(const unsigned long long*)(mbase + l0 + akb);
      float fs[8] = {f0.x, f0.y, f0.z, f0.w, f1.x, f1.y, f1.z, f1.w};
      #pragma unroll
      for (int i = 0; i < 8; ++i) {
        const bool padv = ((mb >> (8 * i)) & 0xffull) != 0;
        a[i] = padv ? (_Float16)0.f
                    : (_Float16)__expf(fs[i] * itau - mrow);
      }
    }
    {
      const float4* sp = (const float4*)(srow + l0 + akb + 8);
      float4 f0 = sp[0], f1 = sp[1];
      unsigned long long mb =
          *(const unsigned long long*)(mbase + l0 + akb + 8);
      float fs[8] = {f0.x, f0.y, f0.z, f0.w, f1.x, f1.y, f1.z, f1.w};
      #pragma unroll
      for (int i = 0; i < 8; ++i) {
        const bool padv = ((mb >> (8 * i)) & 0xffull) != 0;
        a[8 + i] = padv ? (_Float16)0.f
                        : (_Float16)__expf(fs[i] * itau - mrow);
      }
    }

    // ---- B operand: wait for this K-step's DMA, read from LDS ----
    asm volatile("s_wait_asynccnt 0x0" ::: "memory");
    __syncthreads();   // buf[cur] visible to all waves; prior reads of
                       // buf[cur^1] are complete so it can be refilled
    const v16h bm = *(const v16h*)(&sbuf[cur][lan * 32 + bkb]);

    if (l0 + 32 < L_) {  // kick off DMA for the next K-step
      unsigned dst = ldsoff + (unsigned)((cur ^ 1) * 1024);
      unsigned long long src =
          (unsigned long long)(uintptr_t)(vsrc + l0 + 32);
      asm volatile("global_load_async_to_lds_b64 %0, %1, off"
                   :: "v"(dst), "v"(src) : "memory");
    }

    acc = __builtin_amdgcn_wmma_f32_16x16x32_f16(
        false, a, false, bm, (short)0, acc, false, false);
    cur ^= 1;
  }

  #pragma unroll
  for (int j = 0; j < 8; ++j) {
    const int c = c0 + half * 8 + j;
    attn_out[(size_t)(b * C_ + c) * DV + d0 + lan] =
        (_Float16)(acc[j] * rs[j]);
  }
}

// ---------------------------------------------------------------------------
// Kernel 4: out = LN(attn_out @ out_w^T + out_b) * ln_g + ln_b
// ---------------------------------------------------------------------------
__global__ __launch_bounds__(64)
void outln_kernel(const _Float16* __restrict__ attn_out, // (B*C, DV) f16
                  const float* __restrict__ out_w,       // (DV, DV)
                  const float* __restrict__ out_b,
                  const float* __restrict__ ln_g,
                  const float* __restrict__ ln_b,
                  float* __restrict__ out)               // (B*C, DV)
{
  const int r0   = blockIdx.x * 16;
  const int t    = threadIdx.x;
  const int wave = t >> 5;
  const int lane = t & 31;
  const int half = lane >> 4;
  const int lan  = lane & 15;
  const int akb  = half * 16;

  __shared__ float tile[16][DV];
  __shared__ float stats[16][2];

  const _Float16* arow = attn_out + (size_t)(r0 + lan) * DV;

  v8f acc[4] = {};
  #pragma unroll
  for (int k0 = 0; k0 < DV; k0 += 32) {
    v16h a;
    const v8h alo = *(const v8h*)(arow + k0 + akb);
    const v8h ahi = *(const v8h*)(arow + k0 + akb + 8);
    #pragma unroll
    for (int i = 0; i < 8; ++i) { a[i] = alo[i]; a[8 + i] = ahi[i]; }

    #pragma unroll
    for (int nt = 0; nt < 4; ++nt) {
      const int n = (wave * 4 + nt) * 16 + lan;
      const float* wrow = out_w + (size_t)n * DV + k0 + half * 16;
      v16h bm;
      const float4* bp = (const float4*)wrow;
      cvt4(bm, 0, bp[0]); cvt4(bm, 4, bp[1]);
      cvt4(bm, 8, bp[2]); cvt4(bm, 12, bp[3]);
      acc[nt] = __builtin_amdgcn_wmma_f32_16x16x32_f16(
          false, a, false, bm, (short)0, acc[nt], false, false);
    }
  }

  #pragma unroll
  for (int nt = 0; nt < 4; ++nt) {
    const int d = (wave * 4 + nt) * 16 + lan;
    const float bias = out_b[d];
    #pragma unroll
    for (int j = 0; j < 8; ++j)
      tile[half * 8 + j][d] = acc[nt][j] + bias;
  }
  __syncthreads();

  if (t < 16) {
    float su = 0.f, sq = 0.f;
    #pragma unroll 8
    for (int d = 0; d < DV; ++d) {
      const float x = tile[t][d];
      su += x; sq += x * x;
    }
    const float mu  = su * (1.0f / DV);
    const float var = sq * (1.0f / DV) - mu * mu;
    stats[t][0] = mu;
    stats[t][1] = rsqrtf(var + LN_EPS);
  }
  __syncthreads();

  for (int idx = t; idx < 16 * DV; idx += 64) {
    const int rr = idx >> 7;
    const int d  = idx & (DV - 1);
    out[(size_t)(r0 + rr) * DV + d] =
        (tile[rr][d] - stats[rr][0]) * stats[rr][1] * ln_g[d] + ln_b[d];
  }
}

// ---------------------------------------------------------------------------
extern "C" void kernel_launch(void* const* d_in, const int* in_sizes, int n_in,
                              void* d_out, int out_size, void* d_ws, size_t ws_size,
                              hipStream_t stream) {
  const float*         scores  = (const float*)d_in[0];         // (B,C,L)
  const float*         emb     = (const float*)d_in[1];         // (B,L,ED)
  const unsigned char* mask    = (const unsigned char*)d_in[2]; // (B,1,L)
  const float*         w_v     = (const float*)d_in[3];         // (DV,ED)
  const float*         log_tau = (const float*)d_in[4];         // (NH,)
  const float*         out_w   = (const float*)d_in[5];         // (DV,DV)
  const float*         out_b   = (const float*)d_in[6];
  const float*         ln_g    = (const float*)d_in[7];
  const float*         ln_b    = (const float*)d_in[8];
  float*               out     = (float*)d_out;

  char* ws = (char*)d_ws;
  const size_t vt_bytes = (size_t)B_ * DV * L_ * sizeof(_Float16); // 33.5 MB
  _Float16* VT       = (_Float16*)ws;
  float*    rowmax   = (float*)(ws + vt_bytes);
  float*    rowsum   = rowmax + (size_t)B_ * C_ * NH;
  _Float16* attn_out = (_Float16*)(ws + vt_bytes + 65536);

  stats_kernel<<<B_ * C_, 256, 0, stream>>>(scores, mask, log_tau,
                                            rowmax, rowsum);
  vproj_kernel<<<B_ * (L_ / 16), 256, 0, stream>>>(emb, w_v, VT);
  attn_kernel<<<B_ * NH * 2, 128, 0, stream>>>(scores, mask, log_tau, VT,
                                               rowmax, rowsum, attn_out);
  outln_kernel<<<(B_ * C_) / 16, 64, 0, stream>>>(attn_out, out_w, out_b,
                                                  ln_g, ln_b, out);
}
